// BurgersPinn_4294967296648
// MI455X (gfx1250) — matrix operand order repaired and checked by reading, not compiled, not run on hardware
//
#include <hip/hip_runtime.h>

// ---------------------------------------------------------------------------
// Burgers PINN residual on MI455X (gfx1250).
// Forward-mode 2nd-order streams (h, h_t, h_x, h_xx) fused into f16 WMMA
// GEMMs with f32 accumulation. Persistent blocks; per-wave 16-point tiles.
// ---------------------------------------------------------------------------

typedef __attribute__((ext_vector_type(16))) _Float16 v16h;
typedef __attribute__((ext_vector_type(8)))  _Float16 v8h;
typedef __attribute__((ext_vector_type(8)))  float    v8f;

#define N_EQ    131072
#define N_INIT  4096
#define N_BND   4096
#define N_TOT   (N_EQ + N_INIT + N_BND)      // 139264
#define NU_C    0.0031830988618379067f       // 0.01 / pi
#define NCHUNK  (N_TOT / 64)                 // 2176 chunks of 64 points
#define NBLOCKS 544                          // persistent; 4 chunks per block

// ---- LDS layout (bytes) ----------------------------------------------------
#define WT_OFF    0                          // 3 * 128*128 f16, transposed [l][out][in]
#define ACT_OFF   98304                      // 4 waves * 2 bufs * 4 streams * 2048 f16
#define F32_OFF   229376
#define SMEM_BYTES (F32_OFF + 4128)

__device__ __forceinline__ float fast_tanhf(float z) {
  float e = __expf(2.0f * z);
  return 1.0f - 2.0f / (e + 1.0f);
}

// A-fragment (16x32 f16): row m = lane&15, halves 0..7 = K kb..kb+7,
// halves 8..15 = K kb+16..kb+23, kb = k0*32 + (lane>>4)*8.
__device__ __forceinline__ v16h load_a_frag(const _Float16* row, int kb) {
  v8h lo = *(const v8h*)(row + kb);
  v8h hi = *(const v8h*)(row + kb + 16);
  return __builtin_shufflevector(lo, hi, 0, 1, 2, 3, 4, 5, 6, 7,
                                          8, 9, 10, 11, 12, 13, 14, 15);
}

#define WMMA(ACC, A, B) \
  ACC = __builtin_amdgcn_wmma_f32_16x16x32_f16(false, A, false, B, (short)0, ACC, false, false)

extern "C" __global__ __launch_bounds__(128, 1)
void burgers_pinn_wmma(const float* __restrict__ txe,
                       const float* __restrict__ txi,
                       const float* __restrict__ txb,
                       const float* __restrict__ W0g, const float* __restrict__ b0g,
                       const float* __restrict__ W1g, const float* __restrict__ b1g,
                       const float* __restrict__ W2g, const float* __restrict__ b2g,
                       const float* __restrict__ W3g, const float* __restrict__ b3g,
                       const float* __restrict__ W4g, const float* __restrict__ b4g,
                       float* __restrict__ out) {
  extern __shared__ char smem[];
  _Float16* wt  = (_Float16*)(smem + WT_OFF);
  _Float16* act = (_Float16*)(smem + ACT_OFF);
  float*    f   = (float*)(smem + F32_OFF);
  float* w0  = f;            // [2][128]: row0 = dz/dt coeffs, row1 = dz/dx
  float* b0  = f + 256;
  float* bl  = f + 384;      // b1,b2,b3
  float* w4  = f + 768;
  float* b4  = f + 896;
  float* txs = f + 904;      // 64 pts * (t,x)

  const int tid = threadIdx.x;

  // ---- stage weights into LDS once per persistent block -------------------
  {
    const float* Wg[3] = {W1g, W2g, W3g};
    const float* bg[3] = {b1g, b2g, b3g};
#pragma unroll
    for (int l = 0; l < 3; ++l) {
      const float* W = Wg[l];
      _Float16* wl = wt + l * 16384;
      for (int idx = tid; idx < 16384; idx += 128) {
        int k = idx >> 7, n = idx & 127;          // global W is [in k][out n]
        wl[n * 128 + k] = (_Float16)W[idx];       // store transposed [n][k]
      }
      if (tid < 128) bl[l * 128 + tid] = bg[l][tid];
    }
    if (tid < 128) {
      w0[tid]       = W0g[tid];
      w0[128 + tid] = W0g[128 + tid];
      b0[tid]       = b0g[tid];
      w4[tid]       = W4g[tid];
    }
    if (tid == 0) b4[0] = b4g[0];
  }

  const int wave  = tid >> 5;
  const int lane  = tid & 31;
  const int mlane = lane & 15;
  const int hsel  = lane >> 4;                    // 0/1 half-wave select
  _Float16* buf0 = act + wave * 16384;            // [stream][p][n], 4*2048 f16
  _Float16* buf1 = buf0 + 8192;

  // ---- persistent loop over 64-point chunks -------------------------------
  for (int chunk = blockIdx.x; chunk < NCHUNK; chunk += NBLOCKS) {
    // stage this chunk's 64 points (t,x)
    if (tid < 64) {
      int P = chunk * 64 + tid;
      const float* src; int q;
      if (P < N_EQ)              { src = txe; q = P; }
      else if (P < N_EQ + N_INIT){ src = txi; q = P - N_EQ; }
      else                       { src = txb; q = P - N_EQ - N_INIT; }
      txs[tid * 2]     = src[q * 2];
      txs[tid * 2 + 1] = src[q * 2 + 1];
    }
    __syncthreads();

    // ---- layer 0 (2 -> 128): seed h, h_t, h_x, h_xx -----------------------
    for (int idx = lane; idx < 2048; idx += 32) {
      int p = idx >> 7, n = idx & 127;
      float t  = txs[(wave * 16 + p) * 2];
      float x  = txs[(wave * 16 + p) * 2 + 1];
      float ct = w0[n], cx = w0[128 + n];
      float z  = t * ct + x * cx + b0[n];
      float a  = fast_tanhf(z);
      float ap = 1.0f - a * a;
      buf0[idx]        = (_Float16)a;                          // h
      buf0[2048 + idx] = (_Float16)(ap * ct);                  // h_t
      buf0[4096 + idx] = (_Float16)(ap * cx);                  // h_x
      buf0[6144 + idx] = (_Float16)(-2.0f * a * ap * cx * cx); // h_xx
    }

    // ---- hidden layers 1..3 (128 -> 128): 4 streams, n0-pair blocking -----
    for (int l = 0; l < 3; ++l) {
      const _Float16* S  = (l & 1) ? buf1 : buf0;
      _Float16*       D  = (l & 1) ? buf0 : buf1;
      const _Float16* wl = wt + l * 16384;
      const float*    bb = bl + l * 128;

      for (int n0 = 0; n0 < 8; n0 += 2) {
        v8f aA0 = {}; v8f aA1 = {}; v8f aA2 = {}; v8f aA3 = {}; // tile n0
        v8f aB0 = {}; v8f aB1 = {}; v8f aB2 = {}; v8f aB3 = {}; // tile n0+1
#pragma unroll
        for (int k0 = 0; k0 < 4; ++k0) {
          // B fragments (32x16): lane col = mlane, K = k0*32 + hsel*8 + 0..7, +16..23
          const _Float16* wb = wl + mlane * 128 + k0 * 32 + hsel * 8;
          v16h bf0 = load_a_frag(wb + (n0    ) * 16 * 128, 0);
          v16h bf1 = load_a_frag(wb + (n0 + 1) * 16 * 128, 0);
          // A fragments shared by both output tiles (4 streams)
          const int kbA = k0 * 32 + hsel * 8;
          const _Float16* ar = S + mlane * 128;
          v16h f0 = load_a_frag(ar,        kbA);   // h
          v16h f1 = load_a_frag(ar + 2048, kbA);   // h_t
          v16h f2 = load_a_frag(ar + 4096, kbA);   // h_x
          v16h f3 = load_a_frag(ar + 6144, kbA);   // h_xx
          WMMA(aA0, f0, bf0);  WMMA(aB0, f0, bf1);
          WMMA(aA1, f1, bf0);  WMMA(aB1, f1, bf1);
          WMMA(aA2, f2, bf0);  WMMA(aB2, f2, bf1);
          WMMA(aA3, f3, bf0);  WMMA(aB3, f3, bf1);
        }
        // elementwise tanh chain rule; D layout: VGPR r -> row m = r + hsel*8,
        // col n = tile_n0*16 + mlane
        float biasA = bb[(n0    ) * 16 + mlane];
        float biasB = bb[(n0 + 1) * 16 + mlane];
#pragma unroll
        for (int r = 0; r < 8; ++r) {
          int m = r + hsel * 8;
          {
            float z = aA0[r] + biasA, zt = aA1[r], zx = aA2[r], zxx = aA3[r];
            float a = fast_tanhf(z);
            float ap = 1.0f - a * a;
            _Float16* dp = D + m * 128 + n0 * 16 + mlane;
            dp[0]    = (_Float16)a;
            dp[2048] = (_Float16)(ap * zt);
            dp[4096] = (_Float16)(ap * zx);
            dp[6144] = (_Float16)(ap * zxx - 2.0f * a * ap * zx * zx);
          }
          {
            float z = aB0[r] + biasB, zt = aB1[r], zx = aB2[r], zxx = aB3[r];
            float a = fast_tanhf(z);
            float ap = 1.0f - a * a;
            _Float16* dp = D + m * 128 + (n0 + 1) * 16 + mlane;
            dp[0]    = (_Float16)a;
            dp[2048] = (_Float16)(ap * zt);
            dp[4096] = (_Float16)(ap * zx);
            dp[6144] = (_Float16)(ap * zxx - 2.0f * a * ap * zx * zx);
          }
        }
      }
    }

    // ---- output layer (128 -> 1) + Burgers residual -----------------------
    {
      const _Float16* Hf = buf1;                  // after 3 swaps: buf1
      const int p  = mlane;
      const int kb = hsel * 64;
      const v8h* h0 = (const v8h*)(Hf +        p * 128 + kb);
      const v8h* h1 = (const v8h*)(Hf + 2048 + p * 128 + kb);
      const v8h* h2 = (const v8h*)(Hf + 4096 + p * 128 + kb);
      const v8h* h3 = (const v8h*)(Hf + 6144 + p * 128 + kb);
      float su = 0.f, st = 0.f, sx = 0.f, sxx = 0.f;
#pragma unroll
      for (int c = 0; c < 8; ++c) {
        v8h v0 = h0[c], v1 = h1[c], v2 = h2[c], v3 = h3[c];
#pragma unroll
        for (int e = 0; e < 8; ++e) {
          float w = w4[kb + c * 8 + e];
          su  += (float)v0[e] * w;
          st  += (float)v1[e] * w;
          sx  += (float)v2[e] * w;
          sxx += (float)v3[e] * w;
        }
      }
      su  += __shfl_xor(su,  16, 32);
      st  += __shfl_xor(st,  16, 32);
      sx  += __shfl_xor(sx,  16, 32);
      sxx += __shfl_xor(sxx, 16, 32);
      if (lane < 16) {
        int P = chunk * 64 + wave * 16 + p;
        float u = su + b4[0];
        if (P < N_EQ) {
          out[P] = st + u * sx - NU_C * sxx;      // u_t + u*u_x - nu*u_xx
        } else {
          out[P] = u;                             // u at init/boundary points
        }
      }
    }
    __syncthreads();   // guard txs WAR before next chunk's staging
  }
}

extern "C" void kernel_launch(void* const* d_in, const int* in_sizes, int n_in,
                              void* d_out, int out_size, void* d_ws, size_t ws_size,
                              hipStream_t stream) {
  const float* txe = (const float*)d_in[0];
  const float* txi = (const float*)d_in[1];
  const float* txb = (const float*)d_in[2];
  const float* W0  = (const float*)d_in[3];
  const float* b0  = (const float*)d_in[4];
  const float* W1  = (const float*)d_in[5];
  const float* b1  = (const float*)d_in[6];
  const float* W2  = (const float*)d_in[7];
  const float* b2  = (const float*)d_in[8];
  const float* W3  = (const float*)d_in[9];
  const float* b3  = (const float*)d_in[10];
  const float* W4  = (const float*)d_in[11];
  const float* b4  = (const float*)d_in[12];
  float* out = (float*)d_out;

  dim3 grid(NBLOCKS);
  dim3 block(128);
  burgers_pinn_wmma<<<grid, block, SMEM_BYTES, stream>>>(
      txe, txi, txb, W0, b0, W1, b1, W2, b2, W3, b3, W4, b4, out);
}
